// DetDeformableTransformerDecoder_36558761624133
// MI455X (gfx1250) — matrix operand order repaired and checked by reading, not compiled, not run on hardware
//
#include <hip/hip_runtime.h>
#include <cstdint>
#include <cstddef>

// ---------------------------------------------------------------------------
// Model constants
// ---------------------------------------------------------------------------
typedef __bf16 bf16_t;
typedef __attribute__((ext_vector_type(16))) __bf16 v16bf;
typedef __attribute__((ext_vector_type(8)))  float  v8f;

constexpr int C       = 256;
constexpr int NH      = 8;
constexpr int HD      = 32;   // C / NH
constexpr int NL      = 4;
constexpr int NP      = 4;
constexpr int NLAYERS = 6;
constexpr int DFF     = 2048;
constexpr int NQ      = 300;
constexpr int NCLS    = 10;
constexpr int NB      = 2;    // batch
constexpr int L0 = 192, L1 = 96, L2 = 48, L3 = 24;
constexpr int STOT = L0*L0 + L1*L1 + L2*L2 + L3*L3;   // 48960 tokens
constexpr int MQ   = NB * NQ;                          // 600 query rows

// ---------------------------------------------------------------------------
// WMMA fragment layouts (wave32, CDNA5 ISA 7.12.2)
//   A (16x32 bf16): lane m = lane&15, half = lane>>4; elements are two
//       contiguous runs: K = k0+8*half+0..7 and K = k0+16+8*half+0..7
//   B (32x16 bf16): lane n = lane&15, half = lane>>4; elements in the same K order as A
//       -> with N-major packed weights Bt[n*K + k] this is two contiguous 8-element runs
//   C/D (16x16 f32): lane n = lane&15; VGPR r: m = r + 8*(lane>>4)
// ---------------------------------------------------------------------------

// Pack fp32 weight W[K,N] (row-major) into bf16 N-major Bt[n*K + k].
__global__ __launch_bounds__(256)
void pack_bt_kernel(const float* __restrict__ W, bf16_t* __restrict__ Bt,
                    int K, int N)
{
    const int i = blockIdx.x * 256 + threadIdx.x;
    if (i >= K * N) return;
    const int n = i / K;
    const int k = i % K;
    Bt[i] = (bf16_t)W[(size_t)k * N + n];
}

// Generic GEMM: D[M,N] = epilogue(A[M,K] @ B[K,N] + bias)
//   AK = 0: A fp32 row-major (converted in-flight)   AK = 1: A bf16 row-major
//   NT: 16-col tiles per wave (A fragment reused NT times)
//   KC: compile-time K for full unroll (0 = dynamic)
// One wave (32 threads) per 16 x (16*NT) output tile.
template<int AK, int NT, int KC>
__global__ __launch_bounds__(32)
void gemm_wmma_kernel(const void* __restrict__ Av, int lda,
                      const bf16_t* __restrict__ Bt,
                      const float* __restrict__ bias,
                      float* __restrict__ D, int ldd,
                      int M, int N, int K, int op)
{
    const int ngroups = (N + 16*NT - 1) / (16*NT);
    const int tm   = blockIdx.x / ngroups;
    const int tg   = blockIdx.x % ngroups;
    const int row0 = tm * 16;
    const int col0 = tg * 16 * NT;
    const int lane = threadIdx.x & 31;
    const int half = lane >> 4;

    int m = row0 + (lane & 15); if (m >= M) m = M - 1;   // clamp: rows discarded at store

    v8f acc[NT] = {};

    auto kstep = [&](int k0) {
        // ---- A fragment: two contiguous 8-element runs ----
        v16bf a;
        if constexpr (AK == 0) {
            const float* p = (const float*)Av + (size_t)m * lda + k0;
            #pragma unroll
            for (int i = 0; i < 8; ++i) {
                a[i]     = (bf16_t)p[8*half + i];
                a[8 + i] = (bf16_t)p[16 + 8*half + i];
            }
        } else {
            const bf16_t* p = (const bf16_t*)Av + (size_t)m * lda + k0;
            #pragma unroll
            for (int i = 0; i < 8; ++i) {
                a[i]     = p[8*half + i];
                a[8 + i] = p[16 + 8*half + i];
            }
        }
        // ---- NT B fragments (contiguous in packed layout) + NT WMMAs ----
        #pragma unroll
        for (int u = 0; u < NT; ++u) {
            int n = col0 + 16*u + (lane & 15); if (n >= N) n = N - 1;
            const bf16_t* pb = Bt + (size_t)n * K + k0;
            v16bf bvec;
            #pragma unroll
            for (int i = 0; i < 8; ++i) { bvec[i] = pb[8*half + i]; bvec[8 + i] = pb[16 + 8*half + i]; }
            acc[u] = __builtin_amdgcn_wmma_f32_16x16x32_bf16(
                         false, a, false, bvec, (short)0, acc[u], false, false);
        }
    };

    if constexpr (KC > 0) {
        #pragma unroll
        for (int k0 = 0; k0 < KC; k0 += 32) kstep(k0);
    } else {
        for (int k0 = 0; k0 < K; k0 += 32) kstep(k0);
    }

    // ---- epilogue: bias + relu/sigmoid, predicated stores ----
    #pragma unroll
    for (int u = 0; u < NT; ++u) {
        const int n = col0 + 16*u + (lane & 15);
        if (n >= N) continue;
        const float bv = bias ? bias[n] : 0.0f;
        #pragma unroll
        for (int r = 0; r < 8; ++r) {
            const int mm = row0 + r + 8*half;
            if (mm < M) {
                float v = acc[u][r] + bv;
                if (op == 1)      v = fmaxf(v, 0.0f);
                else if (op == 2) v = 1.0f / (1.0f + __expf(-v));
                D[(size_t)mm * ldd + n] = v;
            }
        }
    }
}

// Encoder: src_bf16[b, start+s, d] = bf16( sum_c G[b,c,s]*W[c,d] + bias[d] + pos[d,s] )
// A is the transposed grid view A(m=s,k=c) = G[c*HW + s]; W pre-packed bf16 N-major.
// HW is a compile-time constant so all strided A loads fold into immediate
// instruction offsets (k*HW*4 <= 4.6MB < 24-bit IOFFSET range).
// 16x64 tile per wave; HW is a multiple of 16, N = C = 256 (4 groups of 64).
template<int HW>
__global__ __launch_bounds__(32)
void encoder_wmma_kernel(const float* __restrict__ G,
                         const bf16_t* __restrict__ Wt,
                         const float* __restrict__ bias,
                         const float* __restrict__ pos,
                         bf16_t* __restrict__ src, int startTok)
{
    constexpr int NT = 4;
    const int b    = blockIdx.y;
    const int ngroups = C / (16*NT);     // 4
    const int tm   = blockIdx.x / ngroups;
    const int tg   = blockIdx.x % ngroups;
    const int row0 = tm * 16;
    const int col0 = tg * 16 * NT;
    const int lane = threadIdx.x & 31;
    const int half = lane >> 4;
    const int m    = row0 + (lane & 15);
    // single runtime base per tile; all k offsets are compile-time immediates
    const float* Ga = G + (size_t)b * C * HW + (size_t)(8*half) * HW + m;

    v8f acc[NT] = {};
    #pragma unroll
    for (int k0 = 0; k0 < C; k0 += 32) {
        v16bf a;   // strided gather from channel-major grid (immediate offsets)
        #pragma unroll
        for (int i = 0; i < 8; ++i) {
            a[i]     = (bf16_t)Ga[(size_t)(k0 + i) * HW];
            a[8 + i] = (bf16_t)Ga[(size_t)(k0 + 16 + i) * HW];
        }
        #pragma unroll
        for (int u = 0; u < NT; ++u) {
            const int n = col0 + 16*u + (lane & 15);
            const bf16_t* pb = Wt + (size_t)n * C + k0;
            v16bf bvec;
            #pragma unroll
            for (int i = 0; i < 8; ++i) { bvec[i] = pb[8*half + i]; bvec[8 + i] = pb[16 + 8*half + i]; }
            acc[u] = __builtin_amdgcn_wmma_f32_16x16x32_bf16(
                         false, a, false, bvec, (short)0, acc[u], false, false);
        }
    }

    bf16_t* dst = src + ((size_t)b * STOT + startTok) * C;
    #pragma unroll
    for (int u = 0; u < NT; ++u) {
        const int n = col0 + 16*u + (lane & 15);
        const float bv = bias[n];
        #pragma unroll
        for (int r = 0; r < 8; ++r) {
            const int mm = row0 + r + 8*half;
            const float v = acc[u][r] + bv + pos[(size_t)n * HW + mm];
            dst[(size_t)mm * C + n] = (bf16_t)v;
        }
    }
}

// out[b*NQ+q, :] = query[q, :]
__global__ __launch_bounds__(256)
void init_out_kernel(float* __restrict__ out, const float* __restrict__ query)
{
    const int row = blockIdx.x;
    const int t   = threadIdx.x;
    out[(size_t)row * C + t] = query[(size_t)(row % NQ) * C + t];
}

// Self-attention core: for each (b, h, q): softmax(q.k/sqrt(HD)) . v
// qkv rows: [0:C)=Q(h,d)  [C:2C)=K  [2C:3C)=V
__global__ __launch_bounds__(256)
void mha_attn_kernel(const float* __restrict__ qkv, float* __restrict__ o)
{
    __shared__ float sc[NQ];
    __shared__ float red[256];
    __shared__ float qv[HD];
    __shared__ float pr[256];

    const int q = blockIdx.x, h = blockIdx.y, b = blockIdx.z;
    const int t = threadIdx.x;
    const float scale = 0.17677669529663687f;   // 1/sqrt(32)

    const float* qrow = qkv + ((size_t)(b*NQ + q)) * (3*C) + h*HD;
    if (t < HD) qv[t] = qrow[t];
    __syncthreads();

    for (int k = t; k < NQ; k += 256) {
        const float* krow = qkv + ((size_t)(b*NQ + k)) * (3*C) + C + h*HD;
        float s = 0.f;
        #pragma unroll
        for (int d = 0; d < HD; ++d) s += qv[d] * krow[d];
        sc[k] = s * scale;
    }
    __syncthreads();

    float lm = -1e30f;
    for (int k = t; k < NQ; k += 256) lm = fmaxf(lm, sc[k]);
    red[t] = lm; __syncthreads();
    for (int s2 = 128; s2 > 0; s2 >>= 1) {
        if (t < s2) red[t] = fmaxf(red[t], red[t + s2]);
        __syncthreads();
    }
    const float mx = red[0];
    __syncthreads();

    float ls = 0.f;
    for (int k = t; k < NQ; k += 256) {
        const float e = __expf(sc[k] - mx);
        sc[k] = e; ls += e;
    }
    red[t] = ls; __syncthreads();
    for (int s2 = 128; s2 > 0; s2 >>= 1) {
        if (t < s2) red[t] += red[t + s2];
        __syncthreads();
    }
    const float inv = 1.0f / red[0];
    __syncthreads();

    const int d = t & 31, c = t >> 5;   // 8 chunks x 32 dims
    float p = 0.f;
    for (int k = c; k < NQ; k += 8)
        p += sc[k] * qkv[((size_t)(b*NQ + k)) * (3*C) + 2*C + h*HD + d];
    pr[t] = p; __syncthreads();
    if (t < 32) {
        float s = 0.f;
        #pragma unroll
        for (int cc = 0; cc < 8; ++cc) s += pr[cc*32 + t];
        o[((size_t)(b*NQ + q)) * C + h*HD + t] = s * inv;
    }
}

// out = LayerNorm(out + delta) * g + beta   (one 256-wide row per block)
__global__ __launch_bounds__(256)
void add_ln_kernel(float* __restrict__ out, const float* __restrict__ delta,
                   const float* __restrict__ g, const float* __restrict__ beta)
{
    __shared__ float sh[256];
    __shared__ float stat;
    const int t = threadIdx.x;
    const size_t base = (size_t)blockIdx.x * C;
    const float x = out[base + t] + delta[base + t];

    sh[t] = x; __syncthreads();
    for (int s = 128; s > 0; s >>= 1) { if (t < s) sh[t] += sh[t + s]; __syncthreads(); }
    if (t == 0) stat = sh[0] * (1.0f / C);
    __syncthreads();
    const float mean = stat;
    const float dx = x - mean;
    __syncthreads();
    sh[t] = dx * dx; __syncthreads();
    for (int s = 128; s > 0; s >>= 1) { if (t < s) sh[t] += sh[t + s]; __syncthreads(); }
    if (t == 0) stat = sh[0] * (1.0f / C);
    __syncthreads();
    const float var = stat;
    out[base + t] = dx * rsqrtf(var + 1e-5f) * g[t] + beta[t];
}

// Multi-scale deformable attention sampling.  One wave per (q, h, b); lane = head dim.
__global__ __launch_bounds__(32)
void msda_kernel(const float* __restrict__ off, const float* __restrict__ attl,
                 const float* __restrict__ ref, const float* __restrict__ value,
                 float* __restrict__ o)
{
    const int q = blockIdx.x, h = blockIdx.y, b = blockIdx.z;
    const int lane = threadIdx.x & 31;
    const int DIMS[4] = {L0, L1, L2, L3};

    const float* offp = off  + ((size_t)(b*NQ + q)) * C + h * (NL*NP*2);
    const float* attp = attl + ((size_t)(b*NQ + q)) * (NH*NL*NP) + h * (NL*NP);

    float wt[NL*NP];
    float mx = -1e30f;
    #pragma unroll
    for (int j = 0; j < NL*NP; ++j) { wt[j] = attp[j]; mx = fmaxf(mx, wt[j]); }
    float s = 0.f;
    #pragma unroll
    for (int j = 0; j < NL*NP; ++j) { wt[j] = __expf(wt[j] - mx); s += wt[j]; }
    const float inv = 1.0f / s;

    const float rx = ref[q*2 + 0];
    const float ry = ref[q*2 + 1];

    float acc = 0.f;
    int start = 0;
    #pragma unroll
    for (int l = 0; l < NL; ++l) {
        const int Wd = DIMS[l], Hd = DIMS[l];
        const float* vbase = value + ((size_t)b * STOT + start) * C + h*HD + lane;
        #pragma unroll
        for (int p = 0; p < NP; ++p) {
            const float ox = offp[(l*NP + p)*2 + 0];
            const float oy = offp[(l*NP + p)*2 + 1];
            const float x = (rx + ox / (float)Wd) * (float)Wd - 0.5f;
            const float y = (ry + oy / (float)Hd) * (float)Hd - 0.5f;
            const float x0f = floorf(x), y0f = floorf(y);
            const float wx1 = x - x0f, wx0 = 1.0f - wx1;
            const float wy1 = y - y0f, wy0 = 1.0f - wy1;
            const int x0 = (int)x0f, y0 = (int)y0f;
            const float aw = wt[l*NP + p] * inv;
            float sum = 0.f;
            if (x0   >= 0 && x0   < Wd && y0   >= 0 && y0   < Hd) sum += wx0*wy0 * vbase[(size_t)(y0    *Wd + x0  ) * C];
            if (x0+1 >= 0 && x0+1 < Wd && y0   >= 0 && y0   < Hd) sum += wx1*wy0 * vbase[(size_t)(y0    *Wd + x0+1) * C];
            if (x0   >= 0 && x0   < Wd && y0+1 >= 0 && y0+1 < Hd) sum += wx0*wy1 * vbase[(size_t)((y0+1)*Wd + x0  ) * C];
            if (x0+1 >= 0 && x0+1 < Wd && y0+1 >= 0 && y0+1 < Hd) sum += wx1*wy1 * vbase[(size_t)((y0+1)*Wd + x0+1) * C];
            acc += aw * sum;
        }
        start += Wd * Hd;
    }
    o[((size_t)(b*NQ + q)) * C + h*HD + lane] = acc;
}

// ---------------------------------------------------------------------------
// Host side
// ---------------------------------------------------------------------------
static inline void pack_weight(const float* W, bf16_t* Bt, int K, int N,
                               hipStream_t s)
{
    const int tot = K * N;
    pack_bt_kernel<<<(tot + 255) / 256, 256, 0, s>>>(W, Bt, K, N);
}

// A fp32, NT=4, K=256 fully unrolled
static inline void gemm_f32_k256(const float* A, int lda, const bf16_t* Bt,
                                 const float* bias, float* D, int ldd,
                                 int M, int N, int op, hipStream_t s)
{
    const int ng = (N + 63) / 64;
    gemm_wmma_kernel<0, 4, 256><<<((M + 15) / 16) * ng, 32, 0, s>>>(
        A, lda, Bt, bias, D, ldd, M, N, 256, op);
}
// A bf16, NT=4, K=256 fully unrolled (dominant value GEMM)
static inline void gemm_bf16_k256(const bf16_t* A, int lda, const bf16_t* Bt,
                                  const float* bias, float* D, int ldd,
                                  int M, int N, int op, hipStream_t s)
{
    const int ng = (N + 63) / 64;
    gemm_wmma_kernel<1, 4, 256><<<((M + 15) / 16) * ng, 32, 0, s>>>(
        A, lda, Bt, bias, D, ldd, M, N, 256, op);
}
// A fp32, NT=4, dynamic K (ff2: K=2048)
static inline void gemm_f32_kdyn(const float* A, int lda, const bf16_t* Bt,
                                 const float* bias, float* D, int ldd,
                                 int M, int N, int K, int op, hipStream_t s)
{
    const int ng = (N + 63) / 64;
    gemm_wmma_kernel<0, 4, 0><<<((M + 15) / 16) * ng, 32, 0, s>>>(
        A, lda, Bt, bias, D, ldd, M, N, K, op);
}
// A fp32, NT=1, K=256 (tiny-N heads: N = 2 / 9 / 11)
static inline void gemm_f32_n1(const float* A, int lda, const bf16_t* Bt,
                               const float* bias, float* D, int ldd,
                               int M, int N, int op, hipStream_t s)
{
    gemm_wmma_kernel<0, 1, 256><<<((M + 15) / 16), 32, 0, s>>>(
        A, lda, Bt, bias, D, ldd, M, N, 256, op);
}

extern "C" void kernel_launch(void* const* d_in, const int* in_sizes, int n_in,
                              void* d_out, int out_size, void* d_ws, size_t ws_size,
                              hipStream_t stream)
{
    (void)in_sizes; (void)n_in; (void)out_size; (void)ws_size;

    // ---- unpack inputs (JAX pytree order: dict keys sorted alphabetically) ----
    int idx = 0;
    const float* grids[4];
    for (int l = 0; l < 4; ++l) grids[l] = (const float*)d_in[idx++];

    const float *bbox_b[4], *bbox_w[4];
    for (int i = 0; i < 4; ++i) { bbox_b[i] = (const float*)d_in[idx++];
                                  bbox_w[i] = (const float*)d_in[idx++]; }
    const float* cls_b = (const float*)d_in[idx++];
    const float* cls_w = (const float*)d_in[idx++];

    const float *enc_b[4], *enc_pos[4], *enc_w[4];
    for (int l = 0; l < 4; ++l) {
        enc_b[l]   = (const float*)d_in[idx++];
        enc_pos[l] = (const float*)d_in[idx++];
        enc_w[l]   = (const float*)d_in[idx++];
    }

    struct Layer {
        const float *att_b, *att_w, *ca_b, *ca_w, *ff1_b, *ff1_w, *ff2_b, *ff2_w;
        const float *n1_b, *n1_g, *n2_b, *n2_g, *n3_b, *n3_g;
        const float *off_b, *off_w, *sain_b, *sain_w, *saout_b, *saout_w, *val_b, *val_w;
    } Ls[NLAYERS];
    for (int l = 0; l < NLAYERS; ++l) {
        Layer& L = Ls[l];
        L.att_b   = (const float*)d_in[idx++]; L.att_w   = (const float*)d_in[idx++];
        L.ca_b    = (const float*)d_in[idx++]; L.ca_w    = (const float*)d_in[idx++];
        L.ff1_b   = (const float*)d_in[idx++]; L.ff1_w   = (const float*)d_in[idx++];
        L.ff2_b   = (const float*)d_in[idx++]; L.ff2_w   = (const float*)d_in[idx++];
        L.n1_b    = (const float*)d_in[idx++]; L.n1_g    = (const float*)d_in[idx++];
        L.n2_b    = (const float*)d_in[idx++]; L.n2_g    = (const float*)d_in[idx++];
        L.n3_b    = (const float*)d_in[idx++]; L.n3_g    = (const float*)d_in[idx++];
        L.off_b   = (const float*)d_in[idx++]; L.off_w   = (const float*)d_in[idx++];
        L.sain_b  = (const float*)d_in[idx++]; L.sain_w  = (const float*)d_in[idx++];
        L.saout_b = (const float*)d_in[idx++]; L.saout_w = (const float*)d_in[idx++];
        L.val_b   = (const float*)d_in[idx++]; L.val_w   = (const float*)d_in[idx++];
    }
    const float* query = (const float*)d_in[idx++];
    const float* ref_b = (const float*)d_in[idx++];
    const float* ref_w = (const float*)d_in[idx++];

    // ---- workspace carving ----
    char* wsp = (char*)d_ws;
    auto carve = [&](size_t bytes) -> char* {
        char* p = wsp;
        wsp += (bytes + 255) & ~(size_t)255;
        return p;
    };
    bf16_t* src_bf = (bf16_t*)carve((size_t)NB * STOT * C * sizeof(bf16_t)); // 50 MB, L2-resident
    float*  value  = (float*) carve((size_t)NB * STOT * C * sizeof(float));  // 100 MB
    float*  outq   = (float*) carve((size_t)MQ * C * sizeof(float));
    float*  qkv    = (float*) carve((size_t)MQ * 3 * C * sizeof(float));
    float*  attn_o = (float*) carve((size_t)MQ * C * sizeof(float));
    float*  proj   = (float*) carve((size_t)MQ * C * sizeof(float));
    float*  offb   = (float*) carve((size_t)MQ * C * sizeof(float));
    float*  attb   = (float*) carve((size_t)MQ * (NH*NL*NP) * sizeof(float));
    float*  msda_o = (float*) carve((size_t)MQ * C * sizeof(float));
    float*  hidden = (float*) carve((size_t)MQ * DFF * sizeof(float));
    float*  refb   = (float*) carve((size_t)NQ * 2 * sizeof(float));
    float*  h1     = (float*) carve((size_t)MQ * C * sizeof(float));
    float*  h2     = (float*) carve((size_t)MQ * C * sizeof(float));
    bf16_t* wpack  = (bf16_t*)carve((size_t)DFF * C * sizeof(bf16_t));       // 1 MB pack staging

    float* dout = (float*)d_out;   // [0,6600) classes, [6600,12000) bboxes

    // ---- encoder: project + pos-embed each level into bf16 token buffer ----
    {
        int start = 0;
        // level 0: 192x192
        pack_weight(enc_w[0], wpack, C, C, stream);
        encoder_wmma_kernel<L0*L0><<<dim3((L0*L0/16)*(C/64), NB), 32, 0, stream>>>(
            grids[0], wpack, enc_b[0], enc_pos[0], src_bf, start);
        start += L0*L0;
        // level 1: 96x96
        pack_weight(enc_w[1], wpack, C, C, stream);
        encoder_wmma_kernel<L1*L1><<<dim3((L1*L1/16)*(C/64), NB), 32, 0, stream>>>(
            grids[1], wpack, enc_b[1], enc_pos[1], src_bf, start);
        start += L1*L1;
        // level 2: 48x48
        pack_weight(enc_w[2], wpack, C, C, stream);
        encoder_wmma_kernel<L2*L2><<<dim3((L2*L2/16)*(C/64), NB), 32, 0, stream>>>(
            grids[2], wpack, enc_b[2], enc_pos[2], src_bf, start);
        start += L2*L2;
        // level 3: 24x24
        pack_weight(enc_w[3], wpack, C, C, stream);
        encoder_wmma_kernel<L3*L3><<<dim3((L3*L3/16)*(C/64), NB), 32, 0, stream>>>(
            grids[3], wpack, enc_b[3], enc_pos[3], src_bf, start);
    }

    // ---- reference points: sigmoid(query @ ref_w + ref_b), shared across batch ----
    pack_weight(ref_w, wpack, C, 2, stream);
    gemm_f32_n1(query, C, wpack, ref_b, refb, 2, NQ, 2, /*sigmoid*/2, stream);

    // ---- initialize decoder queries ----
    init_out_kernel<<<MQ, 256, 0, stream>>>(outq, query);

    // ---- decoder layers ----
    for (int l = 0; l < NLAYERS; ++l) {
        const Layer& L = Ls[l];

        // value = src @ W_val + b  (dominant GEMM: 97920 x 256 x 256, bf16 A)
        pack_weight(L.val_w, wpack, C, C, stream);
        gemm_bf16_k256(src_bf, C, wpack, L.val_b, value, C, NB*STOT, C, 0, stream);

        // --- self attention ---
        pack_weight(L.sain_w, wpack, C, 3*C, stream);
        gemm_f32_k256(outq, C, wpack, L.sain_b, qkv, 3*C, MQ, 3*C, 0, stream);
        {
            dim3 grid(NQ, NH, NB);
            mha_attn_kernel<<<grid, 256, 0, stream>>>(qkv, attn_o);
        }
        pack_weight(L.saout_w, wpack, C, C, stream);
        gemm_f32_k256(attn_o, C, wpack, L.saout_b, proj, C, MQ, C, 0, stream);
        add_ln_kernel<<<MQ, 256, 0, stream>>>(outq, proj, L.n2_g, L.n2_b);

        // --- deformable cross attention ---
        pack_weight(L.off_w, wpack, C, C, stream);
        gemm_f32_k256(outq, C, wpack, L.off_b, offb, C, MQ, C, 0, stream);
        pack_weight(L.att_w, wpack, C, NH*NL*NP, stream);
        gemm_f32_k256(outq, C, wpack, L.att_b, attb, NH*NL*NP, MQ, NH*NL*NP, 0, stream);
        {
            dim3 grid(NQ, NH, NB);
            msda_kernel<<<grid, 32, 0, stream>>>(offb, attb, refb, value, msda_o);
        }
        pack_weight(L.ca_w, wpack, C, C, stream);
        gemm_f32_k256(msda_o, C, wpack, L.ca_b, proj, C, MQ, C, 0, stream);
        add_ln_kernel<<<MQ, 256, 0, stream>>>(outq, proj, L.n1_g, L.n1_b);

        // --- FFN ---
        pack_weight(L.ff1_w, wpack, C, DFF, stream);
        gemm_f32_k256(outq, C, wpack, L.ff1_b, hidden, DFF, MQ, DFF, /*relu*/1, stream);
        pack_weight(L.ff2_w, wpack, DFF, C, stream);
        gemm_f32_kdyn(hidden, DFF, wpack, L.ff2_b, proj, C, MQ, C, DFF, 0, stream);
        add_ln_kernel<<<MQ, 256, 0, stream>>>(outq, proj, L.n3_g, L.n3_b);
    }

    // ---- heads ----
    // classes: (600 x 256) @ (256 x 11) + b  -> dout[0 : 6600)
    pack_weight(cls_w, wpack, C, NCLS + 1, stream);
    gemm_f32_n1(outq, C, wpack, cls_b, dout, NCLS + 1, MQ, NCLS + 1, 0, stream);

    // bbox MLP: 3x (256->256, ReLU) then 256->9, sigmoid -> dout[6600 : 12000)
    pack_weight(bbox_w[0], wpack, C, C, stream);
    gemm_f32_k256(outq, C, wpack, bbox_b[0], h1, C, MQ, C, 1, stream);
    pack_weight(bbox_w[1], wpack, C, C, stream);
    gemm_f32_k256(h1,   C, wpack, bbox_b[1], h2, C, MQ, C, 1, stream);
    pack_weight(bbox_w[2], wpack, C, C, stream);
    gemm_f32_k256(h2,   C, wpack, bbox_b[2], h1, C, MQ, C, 1, stream);
    pack_weight(bbox_w[3], wpack, C, 9, stream);
    gemm_f32_n1(h1, C, wpack, bbox_b[3], dout + (size_t)MQ * (NCLS + 1), 9,
                MQ, 9, /*sigmoid*/2, stream);
}
